// EquivariantFeatureEmbeddingNetwork_4140348473568
// MI455X (gfx1250) — matrix operand order repaired and checked by reading, not compiled, not run on hardware
//
#include <hip/hip_runtime.h>

typedef _Float16 h16;
typedef __attribute__((ext_vector_type(16))) _Float16 v16h;
typedef __attribute__((ext_vector_type(8)))  _Float16 v8h;
typedef __attribute__((ext_vector_type(8)))  float    v8f;

#define B_    8
#define D_    3
#define N_    2048
#define KNN_  20
#define P_    (B_ * N_ * KNN_)   /* 327680 edge pixels  */
#define P2_   (B_ * N_)          /* 16384 point pixels  */
#define EMB_  512
#define EPS_  1e-5f

/* Tiled activation layout ("fragment native"):
   element (channel k, pixel p) of a [K][P] matrix is stored at
       ((k/16)*P + p)*16 + (k%16)
   so a WMMA B-fragment (16 consecutive k for one pixel) is 16 contiguous
   halves (32B), and a WMMA C-fragment store is 8 contiguous halves (16B). */
__device__ __forceinline__ size_t tadr(int c, size_t p, size_t P) {
  return ((size_t)(c >> 4) * P + p) * 16 + (size_t)(c & 15);
}

/* ---------------------------------------------------------------- demean */
__global__ void k_demean(const float* __restrict__ pts, float* __restrict__ x) {
  __shared__ float red[256];
  const int row = blockIdx.x;                  // b*3 + d  (24 rows)
  const float* src = pts + (size_t)row * N_;
  float s = 0.f;
  for (int i = threadIdx.x; i < N_; i += 256) s += src[i];
  red[threadIdx.x] = s; __syncthreads();
  for (int o = 128; o > 0; o >>= 1) {
    if (threadIdx.x < o) red[threadIdx.x] += red[threadIdx.x + o];
    __syncthreads();
  }
  const float mean = red[0] * (1.0f / N_);
  float* dst = x + (size_t)row * N_;
  for (int i = threadIdx.x; i < N_; i += 256) dst[i] = src[i] - mean;
}

/* ------------------------------------------------------- kNN top-20 (neg dist) */
__global__ void k_knn(const float* __restrict__ x, int* __restrict__ idx) {
  __shared__ float sx[3 * N_];     // coords of this batch, [d][n]
  __shared__ float dist[N_];
  __shared__ float rv[256];
  __shared__ int   ri[256];
  const int b = blockIdx.x >> 11;          // /2048
  const int n = blockIdx.x & (N_ - 1);
  const float* xb = x + (size_t)b * 3 * N_;
  for (int i = threadIdx.x; i < 3 * N_; i += 256) sx[i] = xb[i];
  __syncthreads();
  const float cx = sx[0 * N_ + n], cy = sx[1 * N_ + n], cz = sx[2 * N_ + n];
  for (int m = threadIdx.x; m < N_; m += 256) {
    float dx = sx[0 * N_ + m] - cx;
    float dy = sx[1 * N_ + m] - cy;
    float dz = sx[2 * N_ + m] - cz;
    dist[m] = -(dx * dx + dy * dy + dz * dz);
  }
  __syncthreads();
  for (int t = 0; t < KNN_; ++t) {
    float bv = -__builtin_inff(); int bi = 0x7fffffff;
    for (int m = threadIdx.x; m < N_; m += 256) {
      float v = dist[m];
      if (v > bv) { bv = v; bi = m; }      // strided ascending -> ties keep smaller m
    }
    rv[threadIdx.x] = bv; ri[threadIdx.x] = bi;
    __syncthreads();
    for (int o = 128; o > 0; o >>= 1) {
      if (threadIdx.x < o) {
        float ov = rv[threadIdx.x + o]; int oi = ri[threadIdx.x + o];
        if (ov > rv[threadIdx.x] || (ov == rv[threadIdx.x] && oi < ri[threadIdx.x])) {
          rv[threadIdx.x] = ov; ri[threadIdx.x] = oi;
        }
      }
      __syncthreads();
    }
    if (threadIdx.x == 0) {
      idx[(size_t)blockIdx.x * KNN_ + t] = ri[0];
      dist[ri[0]] = -__builtin_inff();
    }
    __syncthreads();
  }
}

/* ------------------------------- edge features -> H0 f16 tiled [32][P] */
__global__ void k_build_h0(const float* __restrict__ x, const int* __restrict__ idx,
                           h16* __restrict__ H0) {
  const size_t p = (size_t)blockIdx.x * 256 + threadIdx.x;
  if (p >= P_) return;
  const int bn = (int)(p / KNN_);
  const int n  = bn & (N_ - 1);
  const int b  = bn >> 11;
  const int j  = idx[p];
  const float* xb = x + (size_t)b * 3 * N_;
  v16h lo = {};                         // channels 0..15 for this pixel
  #pragma unroll
  for (int c = 0; c < 3; ++c) {
    lo[c]     = (h16)xb[(size_t)c * N_ + j];   // neighbor feature
    lo[c + 3] = (h16)xb[(size_t)c * N_ + n];   // center feature
  }
  v16h hi = {};                         // channels 16..31 (K padding)
  *(v16h*)(H0 + p * 16)        = lo;    // K16-group 0
  *(v16h*)(H0 + (P_ + p) * 16) = hi;    // K16-group 1
}

/* ----------------------------------------------- pack weights f32 -> f16 padded */
__global__ void k_pack_w(const float* __restrict__ W, h16* __restrict__ Wp,
                         int Cout, int Cin, int Kp) {
  const int i = blockIdx.x * 256 + threadIdx.x;
  if (i >= Cout * Kp) return;
  const int r = i / Kp, c = i % Kp;
  Wp[i] = (c < Cin) ? (h16)W[(size_t)r * Cin + c] : (h16)0.0f;
}

/* ------------------------------------------------------------- WMMA GEMM
   C (tiled f16, [M][P]) = A[M][KP] (row-major f16) x B (tiled f16, [KP][P]).
   4 waves / block, each wave: one 16-col N-tile x four 16-row M-tiles,
   sharing the B fragment across the 4 WMMAs. KP is compile-time so the
   k-loop fully unrolls. A and B fragments both use contiguous k = 16*half + s (wave32). */
template <int KP>
__global__ void k_wmma_gemm(const h16* __restrict__ A, const h16* __restrict__ Bt,
                            h16* __restrict__ C, int P) {
  const int lane   = threadIdx.x & 31;
  const int wave   = threadIdx.x >> 5;
  const int col0   = (blockIdx.x * 4 + wave) * 16;
  const int m0     = blockIdx.y * 64;          // 4 M-tiles per grid.y step
  if (col0 >= P) return;                       // wave-uniform; EXEC stays all-1s
  const int l15    = lane & 15;
  const int half   = lane >> 4;                // half-wave id
  const int akoff  = half * 16;                // A: K 0-15 / 16-31 split
  const int rowoff = half * 8;                 // C: M 0-7 / 8-15 split
  const int col    = col0 + l15;

  v8f acc[4] = {{}, {}, {}, {}};
  #pragma unroll
  for (int k0 = 0; k0 < KP; k0 += 32) {
    /* B fragment: lane holds B[k0+half*16 .. +15][col] = 16 contiguous halves */
    const h16* bp = Bt + ((size_t)(k0 / 16 + half) * P + col) * 16;
    v16h b = *(const v16h*)bp;
    if (k0 + 32 < KP)
      __builtin_prefetch(bp + (size_t)2 * P * 16, 0, 0);   // next K32 step

    #pragma unroll
    for (int mt = 0; mt < 4; ++mt) {
      const h16* ap = A + (size_t)(m0 + mt * 16 + l15) * KP + k0 + akoff;
      v8h alo = *(const v8h*)ap;          // K = k0+akoff    .. +7
      v8h ahi = *(const v8h*)(ap + 8);    // K = k0+8+akoff  .. +7
      v16h a;
      #pragma unroll
      for (int i = 0; i < 8; ++i) { a[i] = alo[i]; a[8 + i] = ahi[i]; }
      acc[mt] = __builtin_amdgcn_wmma_f32_16x16x32_f16(
          /*neg_a=*/false, a, /*neg_b=*/false, b,
          /*c_mod=*/(short)0, acc[mt], /*reuse_a=*/false, /*reuse_b=*/false);
    }
  }
  #pragma unroll
  for (int mt = 0; mt < 4; ++mt) {
    v8h o;
    #pragma unroll
    for (int j = 0; j < 8; ++j) o[j] = (h16)acc[mt][j];
    /* rows m0+mt*16+rowoff .. +7 at pixel col -> 8 contiguous halves (tiled) */
    h16* cp = C + ((size_t)(m0 / 16 + mt) * P + col) * 16 + rowoff;
    *(v8h*)cp = o;
  }
}

/* --------------------------------------------- per-channel BN stats (mean,rsig) */
__global__ void k_bn_stats(const h16* __restrict__ Y, float* __restrict__ stats,
                           long Pn) {
  __shared__ double rs[256], rs2[256];
  const int c = blockIdx.x;
  const h16* base = Y + (size_t)(c >> 4) * Pn * 16 + (c & 15);
  double s = 0.0, s2 = 0.0;
  for (long i = threadIdx.x; i < Pn; i += 256) {
    float v = (float)base[i * 16]; s += v; s2 += (double)v * v;
  }
  rs[threadIdx.x] = s; rs2[threadIdx.x] = s2; __syncthreads();
  for (int o = 128; o > 0; o >>= 1) {
    if (threadIdx.x < o) { rs[threadIdx.x] += rs[threadIdx.x + o];
                           rs2[threadIdx.x] += rs2[threadIdx.x + o]; }
    __syncthreads();
  }
  if (threadIdx.x == 0) {
    float mean = (float)(rs[0] / (double)Pn);
    float var  = (float)(rs2[0] / (double)Pn) - mean * mean;
    stats[2 * c]     = mean;
    stats[2 * c + 1] = rsqrtf(var + EPS_);
  }
}

/* ------------------------------------ BN + ReLU, tiled Y -> tiled H (f16) */
__global__ void k_bn_relu(const h16* __restrict__ Y, const float* __restrict__ stats,
                          const float* __restrict__ g, const float* __restrict__ bt,
                          h16* __restrict__ H, long total, long Pn) {
  const size_t i = (size_t)blockIdx.x * 256 + threadIdx.x;
  if (i >= (size_t)total) return;
  const int c = (int)((i >> 4) / (size_t)Pn) * 16 + (int)(i & 15);
  const float v = ((float)Y[i] - stats[2 * c]) * stats[2 * c + 1] * g[c] + bt[c];
  H[i] = (h16)fmaxf(v, 0.0f);
}

/* ------------------------- max over k of (already-ReLU'd) tiled H -> tiled XC */
__global__ void k_maxpool(const h16* __restrict__ H, h16* __restrict__ XC,
                          int Cnum, int rowoff) {
  const int i = blockIdx.x * 256 + threadIdx.x;
  if (i >= Cnum * P2_) return;
  const int c = i / P2_, bn = i % P2_;
  const h16* base = H + tadr(c, (size_t)bn * KNN_, P_);
  float mx = (float)base[0];
  #pragma unroll
  for (int kk = 1; kk < KNN_; ++kk) mx = fmaxf(mx, (float)base[(size_t)kk * 16]);
  XC[tadr(rowoff + c, (size_t)bn, P2_)] = (h16)mx;
}

/* ------------------- layer-4: BN + ReLU + maxpool fused (no H4 materialized) */
__global__ void k_bn_relu_maxpool(const h16* __restrict__ Y, const float* __restrict__ stats,
                                  const float* __restrict__ g, const float* __restrict__ bt,
                                  h16* __restrict__ XC, int Cnum, int rowoff) {
  const int i = blockIdx.x * 256 + threadIdx.x;
  if (i >= Cnum * P2_) return;
  const int c = i / P2_, bn = i % P2_;
  const float m = stats[2 * c], r = stats[2 * c + 1], gg = g[c], bb = bt[c];
  const h16* base = Y + tadr(c, (size_t)bn * KNN_, P_);
  float mx = 0.0f;  // ReLU floor
  #pragma unroll
  for (int kk = 0; kk < KNN_; ++kk) {
    float v = ((float)base[(size_t)kk * 16] - m) * r * gg + bb;
    mx = fmaxf(mx, v);
  }
  XC[tadr(rowoff + c, (size_t)bn, P2_)] = (h16)mx;
}

/* ------------------------------- final BN + ReLU -> d_out f32 (B, 512, N) */
__global__ void k_bn_relu_out(const h16* __restrict__ Y5, const float* __restrict__ stats,
                              const float* __restrict__ g, const float* __restrict__ bt,
                              float* __restrict__ out) {
  const int i = blockIdx.x * 256 + threadIdx.x;
  if (i >= EMB_ * P2_) return;
  const int c = i / P2_, bn = i % P2_;
  const int b = bn >> 11, n = bn & (N_ - 1);
  const float v = ((float)Y5[tadr(c, (size_t)bn, P2_)] - stats[2 * c]) * stats[2 * c + 1]
                  * g[c] + bt[c];
  out[((size_t)b * EMB_ + c) * N_ + n] = fmaxf(v, 0.0f);
}

/* ------------------------------------------------------------------ launch */
extern "C" void kernel_launch(void* const* d_in, const int* in_sizes, int n_in,
                              void* d_out, int out_size, void* d_ws, size_t ws_size,
                              hipStream_t stream) {
  const float* pts = (const float*)d_in[0];
  const float* W1 = (const float*)d_in[1];
  const float* W2 = (const float*)d_in[2];
  const float* W3 = (const float*)d_in[3];
  const float* W4 = (const float*)d_in[4];
  const float* W5 = (const float*)d_in[5];
  const float* g1 = (const float*)d_in[6];  const float* b1 = (const float*)d_in[7];
  const float* g2 = (const float*)d_in[8];  const float* b2 = (const float*)d_in[9];
  const float* g3 = (const float*)d_in[10]; const float* b3 = (const float*)d_in[11];
  const float* g4 = (const float*)d_in[12]; const float* b4 = (const float*)d_in[13];
  const float* g5 = (const float*)d_in[14]; const float* b5 = (const float*)d_in[15];

  /* workspace bump layout (bytes) */
  constexpr size_t OFF_X    = 0;
  constexpr size_t OFF_IDX  = OFF_X    + (size_t)B_ * 3 * N_ * 4;        // 196608
  constexpr size_t OFF_W1P  = OFF_IDX  + (size_t)P_ * 4;
  constexpr size_t OFF_W2P  = OFF_W1P  + (size_t)64 * 32 * 2;
  constexpr size_t OFF_W3P  = OFF_W2P  + (size_t)64 * 64 * 2;
  constexpr size_t OFF_W4P  = OFF_W3P  + (size_t)128 * 64 * 2;
  constexpr size_t OFF_W5P  = OFF_W4P  + (size_t)256 * 128 * 2;
  constexpr size_t OFF_STAT = OFF_W5P  + (size_t)512 * 512 * 2;
  constexpr size_t OFF_XC   = OFF_STAT + (size_t)512 * 2 * 4;
  constexpr size_t OFF_Y5   = OFF_XC   + (size_t)512 * P2_ * 2;
  constexpr size_t OFF_H0   = OFF_Y5   + (size_t)512 * P2_ * 2;
  constexpr size_t OFF_H1   = OFF_H0   + (size_t)32  * P_ * 2;
  constexpr size_t OFF_H2   = OFF_H1   + (size_t)64  * P_ * 2;
  constexpr size_t OFF_H3   = OFF_H2   + (size_t)64  * P_ * 2;
  constexpr size_t OFF_Y    = OFF_H3   + (size_t)128 * P_ * 2;
  constexpr size_t WS_NEED  = OFF_Y    + (size_t)256 * P_ * 2;           // ~392 MB
  if (ws_size < WS_NEED) return;

  char* ws = (char*)d_ws;
  float* x     = (float*)(ws + OFF_X);
  int*   idx   = (int*)  (ws + OFF_IDX);
  h16*   W1p   = (h16*)  (ws + OFF_W1P);
  h16*   W2p   = (h16*)  (ws + OFF_W2P);
  h16*   W3p   = (h16*)  (ws + OFF_W3P);
  h16*   W4p   = (h16*)  (ws + OFF_W4P);
  h16*   W5p   = (h16*)  (ws + OFF_W5P);
  float* stats = (float*)(ws + OFF_STAT);
  h16*   XC    = (h16*)  (ws + OFF_XC);
  h16*   Y5    = (h16*)  (ws + OFF_Y5);
  h16*   H0    = (h16*)  (ws + OFF_H0);
  h16*   H1    = (h16*)  (ws + OFF_H1);
  h16*   H2    = (h16*)  (ws + OFF_H2);
  h16*   H3    = (h16*)  (ws + OFF_H3);
  h16*   Y     = (h16*)  (ws + OFF_Y);
  float* out   = (float*)d_out;

  /* graph construction */
  k_demean<<<B_ * 3, 256, 0, stream>>>(pts, x);
  k_knn<<<P2_, 256, 0, stream>>>(x, idx);
  k_build_h0<<<(P_ + 255) / 256, 256, 0, stream>>>(x, idx, H0);

  /* weight packing (f32 -> f16, Cin padded to mult. of 32) */
  k_pack_w<<<(64 * 32 + 255) / 256, 256, 0, stream>>>(W1, W1p, 64, 6, 32);
  k_pack_w<<<(64 * 64 + 255) / 256, 256, 0, stream>>>(W2, W2p, 64, 64, 64);
  k_pack_w<<<(128 * 64 + 255) / 256, 256, 0, stream>>>(W3, W3p, 128, 64, 64);
  k_pack_w<<<(256 * 128 + 255) / 256, 256, 0, stream>>>(W4, W4p, 256, 128, 128);
  k_pack_w<<<(512 * 512 + 255) / 256, 256, 0, stream>>>(W5, W5p, 512, 512, 512);

  const int NT = P_ / 16 / 4;   // gemm blocks along pixels (exact: 5120)

  /* layer 1: 6(->32) -> 64 */
  k_wmma_gemm<32><<<dim3(NT, 64 / 64), 128, 0, stream>>>(W1p, H0, Y, P_);
  k_bn_stats<<<64, 256, 0, stream>>>(Y, stats, P_);
  k_bn_relu<<<(int)(((long)64 * P_ + 255) / 256), 256, 0, stream>>>(Y, stats, g1, b1, H1, (long)64 * P_, P_);
  k_maxpool<<<(64 * P2_ + 255) / 256, 256, 0, stream>>>(H1, XC, 64, 0);

  /* layer 2: 64 -> 64 */
  k_wmma_gemm<64><<<dim3(NT, 64 / 64), 128, 0, stream>>>(W2p, H1, Y, P_);
  k_bn_stats<<<64, 256, 0, stream>>>(Y, stats, P_);
  k_bn_relu<<<(int)(((long)64 * P_ + 255) / 256), 256, 0, stream>>>(Y, stats, g2, b2, H2, (long)64 * P_, P_);
  k_maxpool<<<(64 * P2_ + 255) / 256, 256, 0, stream>>>(H2, XC, 64, 64);

  /* layer 3: 64 -> 128 */
  k_wmma_gemm<64><<<dim3(NT, 128 / 64), 128, 0, stream>>>(W3p, H2, Y, P_);
  k_bn_stats<<<128, 256, 0, stream>>>(Y, stats, P_);
  k_bn_relu<<<(int)(((long)128 * P_ + 255) / 256), 256, 0, stream>>>(Y, stats, g3, b3, H3, (long)128 * P_, P_);
  k_maxpool<<<(128 * P2_ + 255) / 256, 256, 0, stream>>>(H3, XC, 128, 128);

  /* layer 4: 128 -> 256, BN+ReLU+maxpool fused */
  k_wmma_gemm<128><<<dim3(NT, 256 / 64), 128, 0, stream>>>(W4p, H3, Y, P_);
  k_bn_stats<<<256, 256, 0, stream>>>(Y, stats, P_);
  k_bn_relu_maxpool<<<(256 * P2_ + 255) / 256, 256, 0, stream>>>(Y, stats, g4, b4, XC, 256, 256);

  /* layer 5: 512 -> 512 on pooled features */
  k_wmma_gemm<512><<<dim3(P2_ / 16 / 4, 512 / 64), 128, 0, stream>>>(W5p, XC, Y5, P2_);
  k_bn_stats<<<512, 256, 0, stream>>>(Y5, stats, P2_);
  k_bn_relu_out<<<(EMB_ * P2_ + 255) / 256, 256, 0, stream>>>(Y5, stats, g5, b5, out);
}
